// Impatient_Reader_Model_74268574482674
// MI455X (gfx1250) — matrix operand order repaired and checked by reading, not compiled, not run on hardware
//
#include <hip/hip_runtime.h>
#include <hip/hip_bf16.h>

// ---------------------------------------------------------------------------
// CDNA5 (gfx1250, wave32) implementation of the "Impatient Reader" model.
// Big GEMMs run on v_wmma_f32_16x16x32_bf16 (bf16 in / f32 accumulate),
// 1x4 register-blocked with double-buffered fragment loads.
// ---------------------------------------------------------------------------

typedef __attribute__((ext_vector_type(16))) __bf16 v16bf;
typedef __attribute__((ext_vector_type(8)))  float  v8f;

#define DIMD 512
#define SC   256
#define SQ   32
#define BB   32

// ---------------------------------------------------------------------------
// Fragment layouts (CDNA5 ISA 7.12.2, wave32):
//  A (16x32 bf16, 8 VGPRs = v16bf per lane):
//    lane L, elem e ->  M = L&15,  K = (e>>3)*16 + (L>>4)*8 + (e&7)
//  B (32x16 bf16):
//    lane L, elem e ->  N = L&15,  K = (e>>3)*16 + (L>>4)*8 + (e&7)
//  C/D (16x16 f32, 8 VGPRs):
//    lane L, vgpr q ->  M = q + (L>>4)*8,  N = L&15
// Packed buffers store fragments contiguously: frag(mt|nt, kt) occupies 512
// elements, lane-major: base + lane*16 + e  (32-byte aligned per lane).
// ---------------------------------------------------------------------------

// Pack an f32 activation matrix [rows x 512] into A-fragment bf16 layout.
// permute==1 maps output row m=b*256+s to source row s*32+b ([S,B,D]->[B,S,D]).
__global__ void pack_a_kernel(const float* __restrict__ src, __bf16* __restrict__ dst,
                              int MT, int KTtot, int koff, int permute) {
    int w = blockIdx.x * 8 + (threadIdx.x >> 5);
    if (w >= MT * 16) return;
    int lane = threadIdx.x & 31;
    int mt = w >> 4, kt = w & 15;
    int M = (mt << 4) + (lane & 15);
    int srow = permute ? ((M & 255) * 32 + (M >> 8)) : M;
    const float* sp = src + (size_t)srow * DIMD;
    v16bf tmp;
#pragma unroll
    for (int e = 0; e < 16; ++e) {
        int K = (kt << 5) + ((e >> 3) << 4) + ((lane >> 4) << 3) + (e & 7);
        tmp[e] = (__bf16)sp[K];
    }
    *((v16bf*)(dst + ((size_t)(mt * KTtot + koff + kt) << 9)) + lane) = tmp;
}

// Pack an f32 weight matrix W[512 k x 512 n] (row-major, k-major) into
// B-fragment bf16 layout (KTtot k-tiles total; this call fills 16 of them).
__global__ void pack_b_kernel(const float* __restrict__ W, __bf16* __restrict__ dst,
                              int KTtot, int koff) {
    int w = blockIdx.x * 8 + (threadIdx.x >> 5);
    if (w >= 32 * 16) return;
    int lane = threadIdx.x & 31;
    int nt = w >> 4, kt = w & 15;
    int N = (nt << 4) + (lane & 15);
    v16bf tmp;
#pragma unroll
    for (int e = 0; e < 16; ++e) {
        int K = (kt << 5) + ((e >> 3) << 4) + ((lane >> 4) << 3) + (e & 7);
        tmp[e] = (__bf16)W[(size_t)K * DIMD + N];
    }
    *((v16bf*)(dst + ((size_t)(nt * KTtot + koff + kt) << 9)) + lane) = tmp;
}

// Generic WMMA GEMM: out[M x 512] = Apack[M x 32*KT] * Bpack + bias1 (+bias2).
// Each wave computes a 16x64 strip (1 M-tile x 4 N-tiles): the A fragment is
// reused by 4 WMMAs, and next-iteration fragments are loaded before the
// current WMMAs issue (double buffering -> partial s_wait_loadcnt).
__global__ void gemm_wmma_kernel(const __bf16* __restrict__ Ap, const __bf16* __restrict__ Bp,
                                 const float* __restrict__ bias1, const float* __restrict__ bias2,
                                 float* __restrict__ out, int MT, int KT) {
    int w = blockIdx.x * 8 + (threadIdx.x >> 5);
    if (w >= MT * 8) return;               // wave-uniform: EXEC stays all-ones for WMMA
    int lane = threadIdx.x & 31;
    int mt = w >> 3, ng = w & 7;           // ng selects a group of 4 N-tiles

    const v16bf* a_base = (const v16bf*)(Ap + ((size_t)mt * KT << 9)) + lane;
    const v16bf* b_base[4];
#pragma unroll
    for (int j = 0; j < 4; ++j)
        b_base[j] = (const v16bf*)(Bp + ((size_t)(ng * 4 + j) * KT << 9)) + lane;

    v8f acc[4] = {{}, {}, {}, {}};

    // prologue: load kt=0 fragments
    v16bf a_cur = a_base[0];
    v16bf b_cur[4];
#pragma unroll
    for (int j = 0; j < 4; ++j) b_cur[j] = b_base[j][0];

    for (int kt = 0; kt < KT - 1; ++kt) {
        // issue next-iteration loads first so they overlap the WMMAs below
        v16bf a_nxt = a_base[(size_t)(kt + 1) * 32];
        v16bf b_nxt[4];
#pragma unroll
        for (int j = 0; j < 4; ++j) b_nxt[j] = b_base[j][(size_t)(kt + 1) * 32];
        if (kt + 8 < KT) __builtin_prefetch(a_base + (size_t)(kt + 8) * 32, 0, 1);

#pragma unroll
        for (int j = 0; j < 4; ++j)
            acc[j] = __builtin_amdgcn_wmma_f32_16x16x32_bf16(false, a_cur, false, b_cur[j],
                                                             (short)0, acc[j], false, false);
        a_cur = a_nxt;
#pragma unroll
        for (int j = 0; j < 4; ++j) b_cur[j] = b_nxt[j];
    }
    // epilogue: last k-step
#pragma unroll
    for (int j = 0; j < 4; ++j)
        acc[j] = __builtin_amdgcn_wmma_f32_16x16x32_bf16(false, a_cur, false, b_cur[j],
                                                         (short)0, acc[j], false, false);

    int row0 = (mt << 4) + ((lane >> 4) << 3);
#pragma unroll
    for (int j = 0; j < 4; ++j) {
        int col  = ((ng * 4 + j) << 4) + (lane & 15);
        float bv = bias1[col];
        if (bias2) bv += bias2[col];
#pragma unroll
        for (int q = 0; q < 8; ++q)
            out[(size_t)(row0 + q) * DIMD + col] = acc[j][q] + bv;
    }
}

// Per-step: rm = r*W_rm + b_rm ; rr_t = tanh(r*W_rr + b_rr). A gathered from f32 r.
__global__ void step_rm_rr_kernel(const float* __restrict__ r,
                                  const __bf16* __restrict__ Wrm, const __bf16* __restrict__ Wrr,
                                  const float* __restrict__ brm, const float* __restrict__ brr,
                                  float* __restrict__ rm_out, float* __restrict__ rr_out) {
    int w = blockIdx.x * 8 + (threadIdx.x >> 5);    // 0..127
    if (w >= 128) return;
    int lane = threadIdx.x & 31;
    int wsel = w >> 6, mt = (w >> 5) & 1, nt = w & 31;
    const __bf16* Bp = wsel ? Wrr : Wrm;
    int M = (mt << 4) + (lane & 15);
    const float* rrow = r + (size_t)M * DIMD;
    v8f acc = {};
    for (int kt = 0; kt < 16; ++kt) {
        v16bf a;
#pragma unroll
        for (int e = 0; e < 16; ++e) {
            int K = (kt << 5) + ((e >> 3) << 4) + ((lane >> 4) << 3) + (e & 7);
            a[e] = (__bf16)rrow[K];
        }
        v16bf b = *((const v16bf*)(Bp + ((size_t)(nt * 16 + kt) << 9)) + lane);
        acc = __builtin_amdgcn_wmma_f32_16x16x32_bf16(false, a, false, b,
                                                      (short)0, acc, false, false);
    }
    int col  = (nt << 4) + (lane & 15);
    int row0 = (mt << 4) + ((lane >> 4) << 3);
    if (!wsel) {
        float bv = brm[col];
#pragma unroll
        for (int q = 0; q < 8; ++q) rm_out[(size_t)(row0 + q) * DIMD + col] = acc[q] + bv;
    } else {
        float bv = brr[col];
#pragma unroll
        for (int q = 0; q < 8; ++q) rr_out[(size_t)(row0 + q) * DIMD + col] = tanhf(acc[q] + bv);
    }
}

// Per-step fused tanh + W_ms matvec: logits[b,s] = sum_d tanh(ctx_dm+rm+qm)*W_ms[d] + b_ms.
// One wave per (b,s) row; wave-level shuffle reduction (wave32).
__global__ void step_logits_kernel(const float* __restrict__ ctx_dm, const float* __restrict__ rm,
                                   const float* __restrict__ qm_all, int t,
                                   const float* __restrict__ W_ms, const float* __restrict__ b_ms,
                                   float* __restrict__ logits) {
    int w = blockIdx.x * 8 + (threadIdx.x >> 5);    // 0..8191
    int lane = threadIdx.x & 31;
    int b = w >> 8, s = w & 255;
    const float* cd  = ctx_dm + (size_t)w * DIMD;
    const float* rmb = rm + (size_t)b * DIMD;
    const float* qmb = qm_all + ((size_t)(t * BB + b)) * DIMD;
    float acc = 0.f;
#pragma unroll 4
    for (int i = 0; i < 16; ++i) {
        int d = i * 32 + lane;
        float x = cd[d] + rmb[d] + qmb[d];
        acc += tanhf(x) * W_ms[d];
    }
    for (int off = 16; off > 0; off >>= 1) acc += __shfl_xor(acc, off, 32);
    if (lane == 0) logits[b * SC + s] = acc + b_ms[0];
}

// Per-step softmax over S=256 + r_new = sum_s p*ctx[b,s,:] + rr_t. One block per b.
__global__ void step_softmax_update_kernel(const float* __restrict__ logits,
                                           const float* __restrict__ ctx,
                                           const float* __restrict__ rr_t,
                                           float* __restrict__ r_io) {
    __shared__ float sm[SC];
    __shared__ float red[SC];
    int b = blockIdx.x, tid = threadIdx.x;
    float l = logits[b * SC + tid];
    red[tid] = l; __syncthreads();
    for (int s = 128; s > 0; s >>= 1) { if (tid < s) red[tid] = fmaxf(red[tid], red[tid + s]); __syncthreads(); }
    float mx = red[0]; __syncthreads();
    float e = __expf(l - mx);
    red[tid] = e; __syncthreads();
    for (int s = 128; s > 0; s >>= 1) { if (tid < s) red[tid] += red[tid + s]; __syncthreads(); }
    float inv = 1.0f / red[0];
    sm[tid] = e * inv; __syncthreads();

    const float* cb = ctx + (size_t)b * SC * DIMD;
    int d0 = tid, d1 = tid + 256;
    float a0 = 0.f, a1 = 0.f;
    for (int s = 0; s < SC; ++s) {
        float p = sm[s];
        const float* row = cb + (size_t)s * DIMD;
        a0 += p * row[d0];
        a1 += p * row[d1];
    }
    r_io[b * DIMD + d0] = a0 + rr_t[b * DIMD + d0];
    r_io[b * DIMD + d1] = a1 + rr_t[b * DIMD + d1];
}

// Final: g = r*W_rg + qh*W_qg + b_rg + b_qg  -> d_out [32 x 512]
__global__ void final_g_kernel(const float* __restrict__ r, const float* __restrict__ qh,
                               const __bf16* __restrict__ Wrg, const __bf16* __restrict__ Wqg,
                               const float* __restrict__ brg, const float* __restrict__ bqg,
                               float* __restrict__ g) {
    int w = blockIdx.x * 8 + (threadIdx.x >> 5);    // 0..63
    if (w >= 64) return;
    int lane = threadIdx.x & 31;
    int mt = (w >> 5) & 1, nt = w & 31;
    int M = (mt << 4) + (lane & 15);
    v8f acc = {};
    for (int pass = 0; pass < 2; ++pass) {
        const float* Arow = (pass ? qh : r) + (size_t)M * DIMD;
        const __bf16* Bp = pass ? Wqg : Wrg;
        for (int kt = 0; kt < 16; ++kt) {
            v16bf a;
#pragma unroll
            for (int e = 0; e < 16; ++e) {
                int K = (kt << 5) + ((e >> 3) << 4) + ((lane >> 4) << 3) + (e & 7);
                a[e] = (__bf16)Arow[K];
            }
            v16bf bfr = *((const v16bf*)(Bp + ((size_t)(nt * 16 + kt) << 9)) + lane);
            acc = __builtin_amdgcn_wmma_f32_16x16x32_bf16(false, a, false, bfr,
                                                          (short)0, acc, false, false);
        }
    }
    int col  = (nt << 4) + (lane & 15);
    float bv = brg[col] + bqg[col];
    int row0 = (mt << 4) + ((lane >> 4) << 3);
#pragma unroll
    for (int q = 0; q < 8; ++q) g[(size_t)(row0 + q) * DIMD + col] = acc[q] + bv;
}

__global__ void zero_f32_kernel(float* p, int n) {
    int i = blockIdx.x * 256 + threadIdx.x;
    if (i < n) p[i] = 0.f;
}

// ---------------------------------------------------------------------------

extern "C" void kernel_launch(void* const* d_in, const int* in_sizes, int n_in,
                              void* d_out, int out_size, void* d_ws, size_t ws_size,
                              hipStream_t stream) {
    (void)in_sizes; (void)n_in; (void)out_size; (void)ws_size;
    const float* ctx_in  = (const float*)d_in[0];   // [256,32,512]
    const float* q_in    = (const float*)d_in[1];   // [32,32,512]
    const float* qh      = (const float*)d_in[2];   // [32,512]
    const float* img_in  = (const float*)d_in[3];   // [256,32,512]
    const float* W_fc1 = (const float*)d_in[4];  const float* b_fc1 = (const float*)d_in[5];
    const float* W_fc2 = (const float*)d_in[6];  const float* b_fc2 = (const float*)d_in[7];
    const float* W_dm  = (const float*)d_in[8];  const float* b_dm  = (const float*)d_in[9];
    const float* W_rm  = (const float*)d_in[10]; const float* b_rm  = (const float*)d_in[11];
    const float* W_qm  = (const float*)d_in[12]; const float* b_qm  = (const float*)d_in[13];
    const float* W_rr  = (const float*)d_in[14]; const float* b_rr  = (const float*)d_in[15];
    const float* W_rg  = (const float*)d_in[16]; const float* b_rg  = (const float*)d_in[17];
    const float* W_qg  = (const float*)d_in[18]; const float* b_qg  = (const float*)d_in[19];
    const float* W_ms  = (const float*)d_in[20]; const float* b_ms  = (const float*)d_in[21];
    float* g_out = (float*)d_out;

    // ---- bump allocator over workspace (256-byte aligned chunks) ----
    char* base = (char*)d_ws;
    size_t off = 0;
    auto alloc = [&](size_t bytes) -> char* {
        char* p = base + off;
        off += (bytes + 255) & ~(size_t)255;
        return p;
    };
    const int M1 = BB * SC;                           // 8192 rows
    __bf16* A1pack   = (__bf16*)alloc((size_t)M1 * 1024 * 2);   // [ctx | img] K=1024
    __bf16* ctxpack  = (__bf16*)alloc((size_t)M1 * DIMD * 2);
    __bf16* qpack    = (__bf16*)alloc((size_t)SQ * BB * DIMD * 2);
    __bf16* W1pack   = (__bf16*)alloc((size_t)1024 * DIMD * 2); // [Wfc1;Wfc2]
    __bf16* Wdmp     = (__bf16*)alloc((size_t)DIMD * DIMD * 2);
    __bf16* Wrmp     = (__bf16*)alloc((size_t)DIMD * DIMD * 2);
    __bf16* Wqmp     = (__bf16*)alloc((size_t)DIMD * DIMD * 2);
    __bf16* Wrrp     = (__bf16*)alloc((size_t)DIMD * DIMD * 2);
    __bf16* Wrgp     = (__bf16*)alloc((size_t)DIMD * DIMD * 2);
    __bf16* Wqgp     = (__bf16*)alloc((size_t)DIMD * DIMD * 2);
    float*  ctx      = (float*)alloc((size_t)M1 * DIMD * 4);
    float*  ctx_dm   = (float*)alloc((size_t)M1 * DIMD * 4);
    float*  qm_all   = (float*)alloc((size_t)SQ * BB * DIMD * 4);
    float*  rm_buf   = (float*)alloc((size_t)BB * DIMD * 4);
    float*  rr_buf   = (float*)alloc((size_t)BB * DIMD * 4);
    float*  r_buf    = (float*)alloc((size_t)BB * DIMD * 4);
    float*  logits   = (float*)alloc((size_t)BB * SC * 4);

    dim3 blk(256);

    // ---- pack weights into B-fragment layout ----
    pack_b_kernel<<<64, blk, 0, stream>>>(W_fc1, W1pack, 32, 0);
    pack_b_kernel<<<64, blk, 0, stream>>>(W_fc2, W1pack, 32, 16);
    pack_b_kernel<<<64, blk, 0, stream>>>(W_dm,  Wdmp, 16, 0);
    pack_b_kernel<<<64, blk, 0, stream>>>(W_rm,  Wrmp, 16, 0);
    pack_b_kernel<<<64, blk, 0, stream>>>(W_qm,  Wqmp, 16, 0);
    pack_b_kernel<<<64, blk, 0, stream>>>(W_rr,  Wrrp, 16, 0);
    pack_b_kernel<<<64, blk, 0, stream>>>(W_rg,  Wrgp, 16, 0);
    pack_b_kernel<<<64, blk, 0, stream>>>(W_qg,  Wqgp, 16, 0);

    // ---- pack activations into A-fragment layout ----
    pack_a_kernel<<<1024, blk, 0, stream>>>(ctx_in, A1pack, 512, 32, 0, 1);   // context half
    pack_a_kernel<<<1024, blk, 0, stream>>>(img_in, A1pack, 512, 32, 16, 1);  // image half
    pack_a_kernel<<<128,  blk, 0, stream>>>(q_in,   qpack,  64, 16, 0, 0);    // all q tokens

    // ---- big GEMMs (1 wave = 16x64 strip; waves = MT*8; 8 waves/block) ----
    // ctx = [context|image] @ [Wfc1;Wfc2] + b_fc1 + b_fc2    (M=8192, K=1024)
    gemm_wmma_kernel<<<512, blk, 0, stream>>>(A1pack, W1pack, b_fc1, b_fc2, ctx, 512, 32);
    // repack ctx for the dm GEMM
    pack_a_kernel<<<1024, blk, 0, stream>>>(ctx, ctxpack, 512, 16, 0, 0);
    // ctx_dm = ctx @ W_dm + b_dm
    gemm_wmma_kernel<<<512, blk, 0, stream>>>(ctxpack, Wdmp, b_dm, (const float*)nullptr,
                                              ctx_dm, 512, 16);
    // qm for all 32 tokens at once (hoisted out of the scan): M=1024
    gemm_wmma_kernel<<<64, blk, 0, stream>>>(qpack, Wqmp, b_qm, (const float*)nullptr,
                                             qm_all, 64, 16);

    // ---- scan over question tokens (sequential dependency through r) ----
    zero_f32_kernel<<<64, blk, 0, stream>>>(r_buf, BB * DIMD);
    for (int t = 0; t < SQ; ++t) {
        step_rm_rr_kernel<<<16, blk, 0, stream>>>(r_buf, Wrmp, Wrrp, b_rm, b_rr,
                                                  rm_buf, rr_buf);
        step_logits_kernel<<<1024, blk, 0, stream>>>(ctx_dm, rm_buf, qm_all, t,
                                                     W_ms, b_ms, logits);
        step_softmax_update_kernel<<<BB, blk, 0, stream>>>(logits, ctx, rr_buf, r_buf);
    }

    // ---- final projection into d_out ----
    final_g_kernel<<<8, blk, 0, stream>>>(r_buf, qh, Wrgp, Wqgp, b_rg, b_qg, g_out);
}
